// QuantMultiheadAttention_12876311953741
// MI455X (gfx1250) — matrix operand order repaired and checked by reading, not compiled, not run on hardware
//
#include <hip/hip_runtime.h>
#include <hip/hip_bf16.h>

typedef __attribute__((ext_vector_type(16))) _Float16 v16h;
typedef __attribute__((ext_vector_type(8)))  float    v8f;

#define NUM_HEADS 12
#define SEQ 1024
#define BATCH 8
#define EMB 768
#define HEAD_DIM 64
#define M_ROWS (SEQ * BATCH)          // 8192
#define N_QKV (3 * EMB)               // 2304
#define OUT_ELEMS ((size_t)M_ROWS * EMB)          // 6291456
#define AVG_ELEMS ((size_t)BATCH * SEQ * SEQ)     // 8388608
#define HEAD_STRIDE (SEQ * HEAD_DIM)  // 65536 halves per head

__device__ __forceinline__ v8f wmma_f16(v16h a, v16h b, v8f c) {
    return __builtin_amdgcn_wmma_f32_16x16x32_f16(false, a, false, b, (short)0, c,
                                                  false, false);
}

// A fragment (16x32 f16, row-major source): lane lm=l&15 is row, half=l>>4.
// v[0..15] = K(16*half + i)
__device__ __forceinline__ v16h load_a_frag(const _Float16* A, int row, int ldk,
                                            int k0, int half) {
    const _Float16* p = A + (size_t)row * ldk + k0;
    v16h a;
#pragma unroll
    for (int i = 0; i < 8; ++i) a[i] = p[16 * half + i];
#pragma unroll
    for (int i = 0; i < 8; ++i) a[8 + i] = p[16 * half + 8 + i];
    return a;
}

// B fragment from pre-packed layout: per (k-step t, column n, half) the 16
// halves a lane needs are contiguous (32B).
__device__ __forceinline__ v16h load_b_packed(const _Float16* Bp, int N, int t,
                                              int n, int half) {
    const _Float16* p = Bp + (((size_t)t * N + n) * 2 + half) * 16;
    v16h b;
#pragma unroll
    for (int e = 0; e < 16; ++e) b[e] = p[e];
    return b;
}

// ---------------- init / reductions / quantization ----------------

__global__ __launch_bounds__(256) void zero_init_kernel(float* avg, size_t n,
                                                        unsigned* scaleSlots) {
    size_t i = (size_t)blockIdx.x * blockDim.x + threadIdx.x;
    size_t stride = (size_t)gridDim.x * blockDim.x;
    for (; i < n; i += stride) avg[i] = 0.0f;
    if (blockIdx.x == 0 && threadIdx.x == 0) { scaleSlots[0] = 0u; scaleSlots[1] = 0u; }
}

__global__ __launch_bounds__(256) void maxabs_kernel(const float* w, int n,
                                                     unsigned* slot) {
    float m = 0.0f;
    for (int i = blockIdx.x * blockDim.x + threadIdx.x; i < n;
         i += gridDim.x * blockDim.x)
        m = fmaxf(m, fabsf(w[i]));
#pragma unroll
    for (int off = 16; off > 0; off >>= 1)
        m = fmaxf(m, __shfl_xor(m, off, 32));
    if ((threadIdx.x & 31) == 0) atomicMax(slot, __float_as_uint(m));
}

// fake-quant W (row-major [N,K]) into packed B-fragment layout:
// out[((k/32)*N + n)*32 + ((k%32)/16)*16 + (k%16)] = fq(W[n*K+k])
__global__ __launch_bounds__(256) void quant_pack_kernel(const float* W, int N,
                                                         int K, const unsigned* slot,
                                                         _Float16* out) {
    int idx = blockIdx.x * blockDim.x + threadIdx.x;
    int total = N * K;
    if (idx >= total) return;
    int n = idx / K, k = idx - n * K;
    float mx = __uint_as_float(*slot);
    float s = mx / 7.0f;
    float w = W[idx];
    float wq = (s > 0.0f) ? fminf(fmaxf(rintf(w / s), -8.0f), 7.0f) * s : 0.0f;
    int t = k >> 5, h = (k >> 4) & 1, e = k & 15;
    out[(((size_t)t * N + n) * 2 + h) * 16 + e] = (_Float16)wq;
}

__global__ __launch_bounds__(256) void cast_f16_kernel(const float* x, _Float16* xh,
                                                       size_t n) {
    size_t i = (size_t)blockIdx.x * blockDim.x + threadIdx.x;
    size_t stride = (size_t)gridDim.x * blockDim.x;
    for (; i < n; i += stride) xh[i] = (_Float16)x[i];
}

// ---------------- QKV projection GEMM (WMMA, 16x128 strip per wave) -------
// C = Xh[8192,768] @ Wq^T -> scatter into head-layout Q/K and packed V.
__global__ __launch_bounds__(32) void gemm_qkv_kernel(const _Float16* Xh,
                                                      const _Float16* Wp,
                                                      const float* bias,
                                                      _Float16* Qh, _Float16* Kh,
                                                      _Float16* Vp) {
    const int K = EMB, N = N_QKV;
    int lane = threadIdx.x & 31, lm = lane & 15, half = lane >> 4;
    int bx = blockIdx.x;
    int mi = bx & 511;          // 512 M tiles
    int ni = bx >> 9;           // 18 N strips of 128
    int m0 = mi * 16, n0 = ni * 128;
    v8f acc[8] = {};
    for (int t = 0; t < K / 32; ++t) {
        // prefetch next k-step tiles toward the WGP while WMMAs drain
        if (t + 1 < K / 32) {
            __builtin_prefetch(Xh + (size_t)(m0 + lm) * K + (t + 1) * 32, 0, 0);
            __builtin_prefetch(Wp + (((size_t)(t + 1) * N + n0 + lane * 4) * 2) * 16, 0, 0);
        }
        v16h a = load_a_frag(Xh, m0 + lm, K, t * 32, half);
#pragma unroll
        for (int j = 0; j < 8; ++j) {
            v16h b = load_b_packed(Wp, N, t, n0 + 16 * j + lm, half);
            acc[j] = wmma_f16(a, b, acc[j]);
        }
    }
#pragma unroll
    for (int j = 0; j < 8; ++j) {
        int n = n0 + 16 * j + lm;
        float bv = bias[n];
#pragma unroll
        for (int r = 0; r < 8; ++r) {
            int m = m0 + r + 8 * half;
            int s = m >> 3, b = m & 7;
            float v = acc[j][r] + bv;
            if (n < EMB) {                         // Q (scaled by D^-0.5)
                int h = n >> 6, d = n & 63;
                Qh[(size_t)(b * NUM_HEADS + h) * HEAD_STRIDE + s * HEAD_DIM + d] =
                    (_Float16)(v * 0.125f);
            } else if (n < 2 * EMB) {              // K, row-major [s][d]
                int nn = n - EMB, h = nn >> 6, d = nn & 63;
                Kh[(size_t)(b * NUM_HEADS + h) * HEAD_STRIDE + s * HEAD_DIM + d] =
                    (_Float16)v;
            } else {                               // V, B-fragment packed by s
                int nn = n - 2 * EMB, h = nn >> 6, d = nn & 63;
                Vp[(size_t)(b * NUM_HEADS + h) * HEAD_STRIDE +
                   ((size_t)(s >> 5) * 64 + d) * 32 + ((s >> 4) & 1) * 16 + (s & 15)] =
                    (_Float16)v;
            }
        }
    }
}

// ---------------- fused attention per (b, h, 16-row tile) ----------------
__global__ __launch_bounds__(32) void attn_kernel(const _Float16* Qh,
                                                  const _Float16* Kh,
                                                  const _Float16* Vp,
                                                  _Float16* Ah, float* avg) {
    __shared__ float sm[16 * SEQ];   // 64 KB score/prob strip
    int lane = threadIdx.x & 31, lm = lane & 15, half = lane >> 4;
    int bid = blockIdx.x;
    int nh = bid >> 6;               // 0..95
    int tile = bid & 63;
    int b = nh / NUM_HEADS, h = nh % NUM_HEADS;
    int s0 = tile * 16;
    const _Float16* Qn = Qh + (size_t)nh * HEAD_STRIDE;
    const _Float16* Kn = Kh + (size_t)nh * HEAD_STRIDE;
    const _Float16* Vn = Vp + (size_t)nh * HEAD_STRIDE;

    // Q A-fragments for both k-steps (K = 64)
    v16h aq0 = load_a_frag(Qn, s0 + lm, HEAD_DIM, 0, half);
    v16h aq1 = load_a_frag(Qn, s0 + lm, HEAD_DIM, 32, half);

    // scores: 64 tiles of 16 columns
    for (int j = 0; j < 64; ++j) {
        int t0 = j * 16;
        if (j + 1 < 64)
            __builtin_prefetch(Kn + (size_t)(t0 + 16 + lm) * HEAD_DIM, 0, 0);
        v8f acc = {};
        {
            v16h bf;
            const _Float16* kp = Kn + (size_t)(t0 + lm) * HEAD_DIM + half * 16;
#pragma unroll
            for (int e = 0; e < 16; ++e) bf[e] = kp[e];
            acc = wmma_f16(aq0, bf, acc);
        }
        {
            v16h bf;
            const _Float16* kp = Kn + (size_t)(t0 + lm) * HEAD_DIM + 32 + half * 16;
#pragma unroll
            for (int e = 0; e < 16; ++e) bf[e] = kp[e];
            acc = wmma_f16(aq1, bf, acc);
        }
#pragma unroll
        for (int r = 0; r < 8; ++r)
            sm[(r + 8 * half) * SEQ + t0 + lm] = acc[r];
    }
    __syncthreads();

    // row-wise softmax over 1024 columns, 32 lanes cooperate per row
    for (int m = 0; m < 16; ++m) {
        float* row = sm + m * SEQ;
        float mx = -3.402823466e+38f;
        for (int t = lane; t < SEQ; t += 32) mx = fmaxf(mx, row[t]);
#pragma unroll
        for (int off = 16; off > 0; off >>= 1)
            mx = fmaxf(mx, __shfl_xor(mx, off, 32));
        float sum = 0.0f;
        for (int t = lane; t < SEQ; t += 32) {
            float e = __expf(row[t] - mx);
            row[t] = e;
            sum += e;
        }
#pragma unroll
        for (int off = 16; off > 0; off >>= 1)
            sum += __shfl_xor(sum, off, 32);
        float inv = 1.0f / sum;
        for (int t = lane; t < SEQ; t += 32) row[t] *= inv;
    }
    __syncthreads();

    // accumulate head-average attention weights
    {
        const float invH = 1.0f / (float)NUM_HEADS;
        float* avgp = avg + (size_t)b * SEQ * SEQ + (size_t)s0 * SEQ;
        for (int idx = lane; idx < 16 * SEQ; idx += 32)
            atomicAdd(avgp + idx, sm[idx] * invH);
    }

    // PV: [16,1024] x [1024,64] with probs from LDS, V pre-packed
    v8f oacc[4] = {};
    for (int kt = 0; kt < 32; ++kt) {
        int k0 = kt * 32;
        if (kt + 1 < 32)
            __builtin_prefetch(Vn + ((size_t)(kt + 1) * 64 + lane * 2) * 32, 0, 0);
        v16h a;
        const float* pr = sm + lm * SEQ + k0;
#pragma unroll
        for (int i = 0; i < 8; ++i) a[i] = (_Float16)pr[16 * half + i];
#pragma unroll
        for (int i = 0; i < 8; ++i) a[8 + i] = (_Float16)pr[16 * half + 8 + i];
#pragma unroll
        for (int j = 0; j < 4; ++j) {
            const _Float16* vp = Vn + ((size_t)kt * 64 + (j * 16 + lm)) * 32 + half * 16;
            v16h bf;
#pragma unroll
            for (int e = 0; e < 16; ++e) bf[e] = vp[e];
            oacc[j] = wmma_f16(a, bf, oacc[j]);
        }
    }
#pragma unroll
    for (int j = 0; j < 4; ++j) {
#pragma unroll
        for (int r = 0; r < 8; ++r) {
            int s = s0 + r + 8 * half;
            int d = j * 16 + lm;
            Ah[((size_t)s * BATCH + b) * EMB + h * HEAD_DIM + d] = (_Float16)oacc[j][r];
        }
    }
}

// ---------------- output projection GEMM (WMMA, 16x128 strip) ----------------
__global__ __launch_bounds__(32) void gemm_out_kernel(const _Float16* Ah,
                                                      const _Float16* Wp,
                                                      const float* bias,
                                                      float* C) {
    const int K = EMB, N = EMB;
    int lane = threadIdx.x & 31, lm = lane & 15, half = lane >> 4;
    int bx = blockIdx.x;
    int mi = bx & 511;
    int ni = bx >> 9;           // 6 strips of 128
    int m0 = mi * 16, n0 = ni * 128;
    v8f acc[8] = {};
    for (int t = 0; t < K / 32; ++t) {
        if (t + 1 < K / 32) {
            __builtin_prefetch(Ah + (size_t)(m0 + lm) * K + (t + 1) * 32, 0, 0);
            __builtin_prefetch(Wp + (((size_t)(t + 1) * N + n0 + lane * 4) * 2) * 16, 0, 0);
        }
        v16h a = load_a_frag(Ah, m0 + lm, K, t * 32, half);
#pragma unroll
        for (int j = 0; j < 8; ++j) {
            v16h b = load_b_packed(Wp, N, t, n0 + 16 * j + lm, half);
            acc[j] = wmma_f16(a, b, acc[j]);
        }
    }
#pragma unroll
    for (int j = 0; j < 8; ++j) {
        int n = n0 + 16 * j + lm;
        float bv = bias[n];
#pragma unroll
        for (int r = 0; r < 8; ++r) {
            int m = m0 + r + 8 * half;
            C[(size_t)m * N + n] = acc[j][r] + bv;
        }
    }
}

extern "C" void kernel_launch(void* const* d_in, const int* in_sizes, int n_in,
                              void* d_out, int out_size, void* d_ws, size_t ws_size,
                              hipStream_t stream) {
    const float* x        = (const float*)d_in[0];   // [1024,8,768]
    const float* w_in     = (const float*)d_in[1];   // [2304,768]
    const float* b_in     = (const float*)d_in[2];   // [2304]
    const float* w_out    = (const float*)d_in[3];   // [768,768]
    const float* b_out    = (const float*)d_in[4];   // [768]

    float* out = (float*)d_out;                      // [1024,8,768]
    float* avg = (float*)d_out + OUT_ELEMS;          // [8,1024,1024]

    char* ws = (char*)d_ws;
    unsigned* scaleSlots = (unsigned*)ws;            // 2 slots, 256B reserved
    _Float16* Wq1 = (_Float16*)(ws + 256);
    _Float16* Wq2 = Wq1 + (size_t)(EMB / 32) * N_QKV * 32;   // 1,769,472 halves
    _Float16* Xh  = Wq2 + (size_t)(EMB / 32) * EMB * 32;     //   589,824 halves
    _Float16* Qh  = Xh + OUT_ELEMS;
    _Float16* Kh  = Qh + (size_t)BATCH * NUM_HEADS * HEAD_STRIDE;
    _Float16* Vp  = Kh + (size_t)BATCH * NUM_HEADS * HEAD_STRIDE;
    _Float16* Ah  = Vp + (size_t)BATCH * NUM_HEADS * HEAD_STRIDE;

    // 1. zero avg-weights output + scale slots
    zero_init_kernel<<<4096, 256, 0, stream>>>(avg, AVG_ELEMS, scaleSlots);

    // 2. per-tensor max-abs for fake quantization
    maxabs_kernel<<<1024, 256, 0, stream>>>(w_in, N_QKV * EMB, scaleSlots + 0);
    maxabs_kernel<<<512, 256, 0, stream>>>(w_out, EMB * EMB, scaleSlots + 1);

    // 3. fake-quant + pack weights into WMMA B-fragment layout
    quant_pack_kernel<<<(N_QKV * EMB + 255) / 256, 256, 0, stream>>>(
        w_in, N_QKV, EMB, scaleSlots + 0, Wq1);
    quant_pack_kernel<<<(EMB * EMB + 255) / 256, 256, 0, stream>>>(
        w_out, EMB, EMB, scaleSlots + 1, Wq2);

    // 4. x -> f16
    cast_f16_kernel<<<4096, 256, 0, stream>>>(x, Xh, OUT_ELEMS);

    // 5. QKV projection (512 M-tiles x 18 N-strips of 128, 1 wave each)
    gemm_qkv_kernel<<<512 * 18, 32, 0, stream>>>(Xh, Wq1, b_in, Qh, Kh, Vp);

    // 6. fused attention: 96 heads x 64 row tiles
    attn_kernel<<<96 * 64, 32, 0, stream>>>(Qh, Kh, Vp, Ah, avg);

    // 7. output projection (512 x 6 strips of 128)
    gemm_out_kernel<<<512 * 6, 32, 0, stream>>>(Ah, Wq2, b_out, out);
}
